// _SelectiveSSM_15590731284548
// MI455X (gfx1250) — compile-verified, matrix operand order checked by reading, not run on hardware
//
#include <hip/hip_runtime.h>
#include <hip/hip_bf16.h>
#include <stdint.h>

// ---------------- problem constants ----------------
#define BATCH    4
#define SEQ      2048
#define D_MODEL  1024
#define D_STATE  16
#define D_CONV   4
#define D_INNER  2048                // D_MODEL * EXPAND
#define MROWS    (BATCH * SEQ)       // 8192

typedef __bf16 bf16;
typedef bf16  bf16x8  __attribute__((ext_vector_type(8)));
typedef bf16  bf16x16 __attribute__((ext_vector_type(16)));
typedef float f32x8   __attribute__((ext_vector_type(8)));

__device__ __forceinline__ bf16 f32_to_bf16(float f) {
    unsigned u = __float_as_uint(f);
    unsigned r = (u + 0x7fffu + ((u >> 16) & 1u)) >> 16;   // round-to-nearest-even
    unsigned short s = (unsigned short)r;
    return __builtin_bit_cast(bf16, s);
}

// async copy global->LDS (CDNA5 GLOBAL_LOAD_ASYNC_TO_LDS_B128, tracked by ASYNCcnt)
__device__ __forceinline__ void async_copy_b128(unsigned lds_byte_addr, const void* gaddr) {
    asm volatile("global_load_async_to_lds_b128 %0, %1, off"
                 :: "v"(lds_byte_addr), "v"(gaddr) : "memory");
}

// ---------------- elementwise converts ----------------
__global__ void cvt_bf16_kernel(const float* __restrict__ in, bf16* __restrict__ out, long n) {
    long i = (long)blockIdx.x * blockDim.x + threadIdx.x;
    if (i < n) out[i] = f32_to_bf16(in[i]);
}

// in: K x N (row-major f32)  ->  out: N x K (row-major bf16)
__global__ void cvt_bf16_T_kernel(const float* __restrict__ in, bf16* __restrict__ out,
                                  int K, int N) {
    long i = (long)blockIdx.x * blockDim.x + threadIdx.x;
    if (i < (long)K * N) {
        int k = (int)(i / N);
        int n = (int)(i % N);
        out[(size_t)n * K + k] = f32_to_bf16(in[i]);
    }
}

// ---------------- WMMA bf16 GEMM: C(f32 MxN) = A(bf16 MxK) * BT(bf16 NxK)^T ------
// Block tile 128x64, staged K-step 64 (2 WMMA K-steps per barrier pair), 256 threads
// (8 waves), each wave owns a 32x32 output = 2x2 WMMA tiles. Double-buffered LDS
// filled by async global->LDS copies (ASYNCcnt), so load latency overlaps compute.
#define TM 128
#define TN 64
#define TK 64
#define LDSP (TK + 8)   // row stride 72 bf16 = 144B (16B aligned)

__global__ __launch_bounds__(256) void gemm_bf16_wmma_kernel(
    const bf16* __restrict__ A, const bf16* __restrict__ BT,
    float* __restrict__ C, int M, int N, int K)
{
    __shared__ __align__(16) bf16 As[2][TM][LDSP];
    __shared__ __align__(16) bf16 Bs[2][TN][LDSP];

    const int tid   = threadIdx.x;
    const int lane  = tid & 31;
    const int wave  = tid >> 5;
    const int waveM = wave & 3;   // 0..3 : 32-row band
    const int waveN = wave >> 2;  // 0..1 : 32-col band
    const int blockM = blockIdx.y * TM;
    const int blockN = blockIdx.x * TN;

    // issue one staged tile (A: 128x64 bf16 -> 4 async b128/thread, B: 64x64 -> 2)
    auto issue_tile = [&](int buf, int k0) {
        #pragma unroll
        for (int j = 0; j < 4; ++j) {
            int idx = tid + j * 256;
            int r   = idx >> 3;            // 0..127
            int cg  = (idx & 7) * 8;       // bf16 col group
            unsigned l = (unsigned)(uintptr_t)&As[buf][r][cg];
            async_copy_b128(l, &A[(size_t)(blockM + r) * K + k0 + cg]);
        }
        #pragma unroll
        for (int j = 0; j < 2; ++j) {
            int idx = tid + j * 256;
            int r   = idx >> 3;            // 0..63
            int cg  = (idx & 7) * 8;
            unsigned l = (unsigned)(uintptr_t)&Bs[buf][r][cg];
            async_copy_b128(l, &BT[(size_t)(blockN + r) * K + k0 + cg]);
        }
    };

    f32x8 acc[2][2];
    acc[0][0] = {}; acc[0][1] = {}; acc[1][0] = {}; acc[1][1] = {};

    const int ntiles = K / TK;
    issue_tile(0, 0);

    const int mrow0 = waveM * 32 + (lane & 15);
    const int nrow0 = waveN * 32 + (lane & 15);
    const int kb    = (lane < 16) ? 0 : 8;   // ISA 16-bit operand K-half split at lane 16

    for (int t = 0; t < ntiles; ++t) {
        const int buf = t & 1;
        if (t + 1 < ntiles) {
            issue_tile(buf ^ 1, (t + 1) * TK);
            asm volatile("s_wait_asynccnt 0x6" ::: "memory");  // tile t's 6 copies done
        } else {
            asm volatile("s_wait_asynccnt 0x0" ::: "memory");
        }
        __syncthreads();   // tile t visible to all waves

        #pragma unroll
        for (int kk = 0; kk < TK; kk += 32) {
            bf16x16 afrag[2], bfrag[2];
            #pragma unroll
            for (int i = 0; i < 2; ++i) {
                bf16x8 lo = *(const bf16x8*)&As[buf][mrow0 + i * 16][kk + kb];
                bf16x8 hi = *(const bf16x8*)&As[buf][mrow0 + i * 16][kk + kb + 16];
                #pragma unroll
                for (int e = 0; e < 8; ++e) { afrag[i][e] = lo[e]; afrag[i][e + 8] = hi[e]; }
                bf16x8 blo = *(const bf16x8*)&Bs[buf][nrow0 + i * 16][kk + kb];
                bf16x8 bhi = *(const bf16x8*)&Bs[buf][nrow0 + i * 16][kk + kb + 16];
                #pragma unroll
                for (int e = 0; e < 8; ++e) { bfrag[i][e] = blo[e]; bfrag[i][e + 8] = bhi[e]; }
            }
            #pragma unroll
            for (int mi = 0; mi < 2; ++mi)
                #pragma unroll
                for (int ni = 0; ni < 2; ++ni)
                    acc[mi][ni] = __builtin_amdgcn_wmma_f32_16x16x32_bf16(
                        false, afrag[mi], false, bfrag[ni],
                        (short)0, acc[mi][ni], false, false);
        }
        __syncthreads();   // all waves done reading buf -> safe to refill next iter
    }

    // epilogue: C/D layout — VGPR r holds M=r (lanes 0-15) / M=r+8 (lanes 16-31), N=lane&15
    const int mb  = (lane >> 4) * 8;
    const int col = lane & 15;
    #pragma unroll
    for (int mi = 0; mi < 2; ++mi)
        #pragma unroll
        for (int ni = 0; ni < 2; ++ni) {
            int m0 = blockM + waveM * 32 + mi * 16 + mb;
            int n0 = blockN + waveN * 32 + ni * 16 + col;
            #pragma unroll
            for (int r = 0; r < 8; ++r)
                C[(size_t)(m0 + r) * N + n0] = acc[mi][ni][r];
        }
}

// ---------------- causal depthwise conv (K=4) + bias + SiLU ----------------
// xz: (B*L, 2*D_INNER) f32; writes xc f32 and xc bf16, layout (B*L, D_INNER)
__global__ void conv_silu_kernel(const float* __restrict__ xz,
                                 const float* __restrict__ conv_w,
                                 const float* __restrict__ conv_b,
                                 float* __restrict__ xc, bf16* __restrict__ xcb)
{
    long i = (long)blockIdx.x * blockDim.x + threadIdx.x;   // over B*L*D_INNER
    if (i >= (long)MROWS * D_INNER) return;
    int  d  = (int)(i % D_INNER);
    long bl = i / D_INNER;
    int  l  = (int)(bl % SEQ);
    long b  = bl / SEQ;

    float acc = conv_b[d];
    const float* base = xz + (size_t)b * SEQ * (2 * D_INNER) + d;
    #pragma unroll
    for (int k = 0; k < D_CONV; ++k) {
        int ls = l - (D_CONV - 1) + k;
        if (ls >= 0) acc += conv_w[d * D_CONV + k] * base[(size_t)ls * (2 * D_INNER)];
    }
    float s = acc / (1.0f + __expf(-acc));   // SiLU
    xc[i]  = s;
    xcb[i] = f32_to_bf16(s);
}

// ---------------- bdt = xc @ W_x   (W_x: D_INNER x 32) ----------------
__global__ __launch_bounds__(256) void bdt_kernel(const float* __restrict__ xc,
                                                  const float* __restrict__ Wx,
                                                  float* __restrict__ bdt)
{
    __shared__ float red[8][32];
    const int row = blockIdx.x;          // 0..MROWS-1
    const int n   = threadIdx.x & 31;
    const int g   = threadIdx.x >> 5;    // 0..7
    const int KS  = D_INNER / 8;         // 256

    const float* xr = xc + (size_t)row * D_INNER;
    float acc = 0.f;
    for (int k = g * KS; k < (g + 1) * KS; ++k)
        acc += xr[k] * Wx[(size_t)k * 32 + n];
    red[g][n] = acc;
    __syncthreads();
    if (g == 0) {
        float s = 0.f;
        #pragma unroll
        for (int j = 0; j < 8; ++j) s += red[j][n];
        bdt[(size_t)row * 32 + n] = s;
    }
}

// ---------------- dt = softplus(dt_pre + b_dt), in place ----------------
__global__ void softplus_kernel(float* __restrict__ dt, const float* __restrict__ b_dt) {
    long i = (long)blockIdx.x * blockDim.x + threadIdx.x;
    if (i >= (long)MROWS * D_INNER) return;
    float v = dt[i] + b_dt[i % D_INNER];
    dt[i] = (v > 20.f) ? v : log1pf(__expf(v));
}

// ---------------- selective scan, fused D-skip + silu(z) gate -------------
// one thread per (b, d) channel: h[16] in registers, sequential over L
__global__ __launch_bounds__(256) void scan_kernel(
    const float* __restrict__ dt, const float* __restrict__ xc,
    const float* __restrict__ xz, const float* __restrict__ bdt,
    const float* __restrict__ A_log, const float* __restrict__ D_param,
    bf16* __restrict__ yb)
{
    int t = blockIdx.x * blockDim.x + threadIdx.x;
    if (t >= BATCH * D_INNER) return;
    const int b = t / D_INNER;
    const int d = t % D_INNER;

    float A[D_STATE], h[D_STATE];
    #pragma unroll
    for (int n = 0; n < D_STATE; ++n) {
        A[n] = -__expf(A_log[(size_t)d * D_STATE + n]);
        h[n] = 0.f;
    }
    const float Dp = D_param[d];

    for (int l = 0; l < SEQ; ++l) {
        const size_t blo = (size_t)b * SEQ + l;
        const float dtv = dt[blo * D_INNER + d];
        const float xv  = xc[blo * D_INNER + d];
        const float zv  = xz[blo * (2 * D_INNER) + D_INNER + d];
        const float* bc = bdt + blo * 32;           // [0..15]=B_p, [16..31]=C_p (uniform addr)
        const float dtx = dtv * xv;

        float y = 0.f;
        #pragma unroll
        for (int n = 0; n < D_STATE; ++n) {
            float dA = __expf(dtv * A[n]);
            h[n] = dA * h[n] + dtx * bc[n];
            y   += h[n] * bc[16 + n];
        }
        float yt = y + xv * Dp;
        float og = yt * (zv / (1.0f + __expf(-zv)));   // * silu(z)
        yb[blo * D_INNER + d] = f32_to_bf16(og);
    }
}

// ---------------- host side ----------------
extern "C" void kernel_launch(void* const* d_in, const int* in_sizes, int n_in,
                              void* d_out, int out_size, void* d_ws, size_t ws_size,
                              hipStream_t stream)
{
    (void)in_sizes; (void)n_in; (void)out_size; (void)ws_size;

    const float* x      = (const float*)d_in[0];   // (B, L, D_MODEL)
    const float* W_in   = (const float*)d_in[1];   // (D_MODEL, 2*D_INNER)
    const float* conv_w = (const float*)d_in[2];   // (D_INNER, 1, 4)
    const float* conv_b = (const float*)d_in[3];   // (D_INNER)
    const float* W_x    = (const float*)d_in[4];   // (D_INNER, 32)
    const float* W_dt   = (const float*)d_in[5];   // (D_INNER, D_INNER)
    const float* b_dt   = (const float*)d_in[6];   // (D_INNER)
    const float* W_out  = (const float*)d_in[7];   // (D_INNER, D_MODEL)
    const float* A_log  = (const float*)d_in[8];   // (D_INNER, 16)
    const float* D_par  = (const float*)d_in[9];   // (D_INNER)
    float* out = (float*)d_out;                    // (B, L, D_MODEL)

    // workspace carve-up (256B aligned)
    char*  ws  = (char*)d_ws;
    size_t off = 0;
    auto alloc = [&](size_t bytes) -> char* {
        char* p = ws + off;
        off += (bytes + 255) & ~size_t(255);
        return p;
    };
    bf16*  x_bf   = (bf16*) alloc((size_t)MROWS * D_MODEL * 2);        // 16 MB
    bf16*  WinT   = (bf16*) alloc((size_t)(2*D_INNER) * D_MODEL * 2);  //  8 MB
    bf16*  WdtT   = (bf16*) alloc((size_t)D_INNER * D_INNER * 2);      //  8 MB
    bf16*  WoutT  = (bf16*) alloc((size_t)D_MODEL * D_INNER * 2);      //  4 MB
    float* xz     = (float*)alloc((size_t)MROWS * (2*D_INNER) * 4);    // 128 MB
    float* xc     = (float*)alloc((size_t)MROWS * D_INNER * 4);        // 64 MB
    bf16*  xc_bf  = (bf16*) alloc((size_t)MROWS * D_INNER * 2);        // 32 MB
    float* bdt    = (float*)alloc((size_t)MROWS * 32 * 4);             //  1 MB
    float* dt     = (float*)alloc((size_t)MROWS * D_INNER * 4);        // 64 MB
    bf16*  y_bf   = (bf16*) alloc((size_t)MROWS * D_INNER * 2);        // 32 MB

    const int T = 256;
    // 1) precision casts / weight transposes
    {
        long n = (long)MROWS * D_MODEL;
        cvt_bf16_kernel<<<(n + T - 1) / T, T, 0, stream>>>(x, x_bf, n);
    }
    {
        long n = (long)D_MODEL * (2*D_INNER);
        cvt_bf16_T_kernel<<<(n + T - 1) / T, T, 0, stream>>>(W_in, WinT, D_MODEL, 2*D_INNER);
    }
    {
        long n = (long)D_INNER * D_INNER;
        cvt_bf16_T_kernel<<<(n + T - 1) / T, T, 0, stream>>>(W_dt, WdtT, D_INNER, D_INNER);
    }
    {
        long n = (long)D_INNER * D_MODEL;
        cvt_bf16_T_kernel<<<(n + T - 1) / T, T, 0, stream>>>(W_out, WoutT, D_INNER, D_MODEL);
    }

    // 2) xz = x @ W_in          (M=8192, N=4096, K=1024)
    gemm_bf16_wmma_kernel<<<dim3((2*D_INNER)/TN, MROWS/TM), T, 0, stream>>>(
        x_bf, WinT, xz, MROWS, 2*D_INNER, D_MODEL);

    // 3) causal depthwise conv + SiLU
    {
        long n = (long)MROWS * D_INNER;
        conv_silu_kernel<<<(n + T - 1) / T, T, 0, stream>>>(xz, conv_w, conv_b, xc, xc_bf);
    }

    // 4) bdt = xc @ W_x  (skinny N=32 -> VALU kernel)
    bdt_kernel<<<MROWS, T, 0, stream>>>(xc, W_x, bdt);

    // 5) dt_pre = xc @ W_dt     (M=8192, N=2048, K=2048)
    gemm_bf16_wmma_kernel<<<dim3(D_INNER/TN, MROWS/TM), T, 0, stream>>>(
        xc_bf, WdtT, dt, MROWS, D_INNER, D_INNER);

    // 6) dt = softplus(dt_pre + b_dt)
    {
        long n = (long)MROWS * D_INNER;
        softplus_kernel<<<(n + T - 1) / T, T, 0, stream>>>(dt, b_dt);
    }

    // 7) selective scan + D-skip + silu(z) gate -> bf16
    scan_kernel<<<(BATCH * D_INNER + T - 1) / T, T, 0, stream>>>(
        dt, xc, xz, bdt, A_log, D_par, y_bf);

    // 8) out = y_gated @ W_out  (M=8192, N=1024, K=2048)
    gemm_bf16_wmma_kernel<<<dim3(D_MODEL/TN, MROWS/TM), T, 0, stream>>>(
        y_bf, WoutT, out, MROWS, D_MODEL, D_INNER);
}
